// ELOTLoss_50826642981485
// MI455X (gfx1250) — matrix operand order repaired and checked by reading, not compiled, not run on hardware
//
#include <hip/hip_runtime.h>
#include <hip/hip_bf16.h>

// ---------------------------------------------------------------------------
// ELOT loss (warmup branch) for MI455X / gfx1250, wave32 + WMMA bf16.
//   - All GEMM work on v_wmma_f32_16x16x32_bf16; each wave owns a 16x64
//     output strip (4 accumulators, A-fragment reused 4x).
//   - fp32->bf16 via native __bf16 conversion (v_cvt path, not bit-twiddle).
//   - RBF kernel matrices never materialized: exp + one-hot class scatter
//     fused into the distance-tile epilogue.
//   - Sinkhorn: 2000 dependent matvec launches; Kmat & Kmat^T L2-resident.
// ---------------------------------------------------------------------------

typedef __attribute__((ext_vector_type(16))) __bf16        bf16x16;
typedef __attribute__((ext_vector_type(8)))  float         f32x8;
typedef __attribute__((ext_vector_type(4)))  float         f32x4;
typedef __attribute__((ext_vector_type(4)))  unsigned int  u32x4;

union ABFrag {
    bf16x16 v;
    __bf16  e[16];
    u32x4   q[2];
};

__device__ __forceinline__ __bf16 f2bf(float f)  { return static_cast<__bf16>(f); }
__device__ __forceinline__ float  bf2f(__bf16 b) { return static_cast<float>(b); }

// A fragment (16x32 bf16, ISA 7.12.2): lane m=l&15, klo=(l>>4)*8;
// slots 0..7  <- k = klo..klo+7 ; slots 8..15 <- k = 16+klo..16+klo+7
__device__ __forceinline__ void load_afrag_bf16(ABFrag& a, const __bf16* A, int ld,
                                                int row, int kb, int klo) {
    const __bf16* p = A + (size_t)row * ld + kb + klo;
    a.q[0] = *reinterpret_cast<const u32x4*>(p);
    a.q[1] = *reinterpret_cast<const u32x4*>(p + 16);
}
// B fragment (32x16 bf16): lane n=l&15 (row of Y for X@Y^T), hi=l>>4;
// slots 0..7 <- k = kb + hi*8 + s ; slots 8..15 <- k = kb + 16 + hi*8 + (s-8)
__device__ __forceinline__ void load_bfrag_bf16(ABFrag& b, const __bf16* B, int ld,
                                                int row, int kb, int hi) {
    const __bf16* p = B + (size_t)row * ld + kb + hi * 8;
    b.q[0] = *reinterpret_cast<const u32x4*>(p);
    b.q[1] = *reinterpret_cast<const u32x4*>(p + 16);
}

__device__ __forceinline__ f32x8 wmma_bf16(const ABFrag& a, const ABFrag& b, f32x8 c) {
    return __builtin_amdgcn_wmma_f32_16x16x32_bf16(false, a.v, false, b.v,
                                                   (short)0, c, false, false);
}

// ---------------------------------------------------------------------------
// P[Mpad x 512] (bf16) = X[M x Kd] (f32) @ W[512 x Kd]^T. fp32->bf16 native
// converts on load. One wave -> 16x64 strip (4 N-tiles). Rows >= M zeroed.
// ---------------------------------------------------------------------------
__global__ void elot_proj_wmma(const float* __restrict__ X, const float* __restrict__ W,
                               __bf16* __restrict__ P, int M, int Mpad, int Kd) {
    int wid  = (blockIdx.x * blockDim.x + threadIdx.x) >> 5;
    int lane = threadIdx.x & 31;
    const int NG = 8;                        // 512 / 64 column groups
    int ntiles = (Mpad >> 4) * NG;
    if (wid >= ntiles) return;
    int ti = wid / NG, tg = wid % NG;
    int m = lane & 15, hi = lane >> 4, klo = hi * 8;
    int arow = ti * 16 + m;
    f32x8 c[4];
    #pragma unroll
    for (int q = 0; q < 4; ++q) c[q] = f32x8{0.f, 0.f, 0.f, 0.f, 0.f, 0.f, 0.f, 0.f};
    for (int kb = 0; kb < Kd; kb += 32) {
        ABFrag a;
        if (arow < M) {
            const float* pa = X + (size_t)arow * Kd + kb + klo;
            f32x4 a0 = *reinterpret_cast<const f32x4*>(pa);
            f32x4 a1 = *reinterpret_cast<const f32x4*>(pa + 4);
            f32x4 a2 = *reinterpret_cast<const f32x4*>(pa + 16);
            f32x4 a3 = *reinterpret_cast<const f32x4*>(pa + 20);
            #pragma unroll
            for (int t = 0; t < 4; ++t) {
                a.e[t]      = f2bf(a0[t]);  a.e[4 + t]  = f2bf(a1[t]);
                a.e[8 + t]  = f2bf(a2[t]);  a.e[12 + t] = f2bf(a3[t]);
            }
        } else {
            #pragma unroll
            for (int t = 0; t < 16; ++t) a.e[t] = f2bf(0.0f);
        }
        #pragma unroll
        for (int q = 0; q < 4; ++q) {
            int brow = (tg * 4 + q) * 16 + m;          // W row (n = m)
            const float* pb = W + (size_t)brow * Kd + kb + hi * 8;
            f32x4 b0 = *reinterpret_cast<const f32x4*>(pb);
            f32x4 b1 = *reinterpret_cast<const f32x4*>(pb + 4);
            f32x4 b2 = *reinterpret_cast<const f32x4*>(pb + 16);
            f32x4 b3 = *reinterpret_cast<const f32x4*>(pb + 20);
            ABFrag b;
            #pragma unroll
            for (int t = 0; t < 4; ++t) {
                b.e[t]      = f2bf(b0[t]);  b.e[4 + t]  = f2bf(b1[t]);
                b.e[8 + t]  = f2bf(b2[t]);  b.e[12 + t] = f2bf(b3[t]);
            }
            c[q] = wmma_bf16(a, b, c[q]);
        }
    }
    // C/D layout: VGPR r -> M = r + 8*hi, N = lane&15
    #pragma unroll
    for (int q = 0; q < 4; ++q) {
        #pragma unroll
        for (int r = 0; r < 8; ++r) {
            int orow = ti * 16 + r + hi * 8;
            int ocol = (tg * 4 + q) * 16 + (lane & 15);
            P[(size_t)orow * 512 + ocol] = f2bf(c[q][r]);
        }
    }
}

// Row squared-norms of bf16 P[rows x 512]; one wave per row.
__global__ void elot_rownorm(const __bf16* __restrict__ P, float* __restrict__ nrm, int rows) {
    int w = (blockIdx.x * blockDim.x + threadIdx.x) >> 5;
    int lane = threadIdx.x & 31;
    if (w >= rows) return;
    const __bf16* p = P + (size_t)w * 512;
    float s = 0.f;
    for (int i = lane; i < 512; i += 32) { float x = bf2f(p[i]); s += x * x; }
    #pragma unroll
    for (int off = 16; off; off >>= 1) s += __shfl_down(s, off, 32);
    if (lane == 0) nrm[w] = s;
}

__global__ void elot_count(const int* __restrict__ tgt, float* __restrict__ cnt, int n) {
    int i = blockIdx.x * blockDim.x + threadIdx.x;
    if (i < n) atomicAdd(&cnt[tgt[i]], 1.0f);
}

// ---------------------------------------------------------------------------
// Fused: 16x64 strip of exp(-0.5*||PA_i - PB_j||^2) over K=512 via WMMA,
// epilogue scatters into class sums (mode 0/1: diagonal-class s_tt/s_ss,
// mode 2: full S_ts[1000x1000]). Kernel matrices never hit memory.
// ---------------------------------------------------------------------------
__global__ void elot_dist_cls(const __bf16* __restrict__ PA, const __bf16* __restrict__ PB,
                              const float* __restrict__ an, const float* __restrict__ bn,
                              const int* __restrict__ tgt, float* __restrict__ out,
                              int mode) {
    int wid  = (blockIdx.x * blockDim.x + threadIdx.x) >> 5;
    int lane = threadIdx.x & 31;
    const int TI = 256, NG = 64;             // 4096/16 row tiles, 4096/64 col groups
    if (wid >= TI * NG) return;
    int ti = wid / NG, tg = wid % NG;
    int m = lane & 15, hi = lane >> 4, klo = hi * 8;
    int arow = ti * 16 + m;
    f32x8 c[4];
    #pragma unroll
    for (int q = 0; q < 4; ++q) c[q] = f32x8{0.f, 0.f, 0.f, 0.f, 0.f, 0.f, 0.f, 0.f};
    #pragma unroll 2
    for (int kb = 0; kb < 512; kb += 32) {
        ABFrag a;
        load_afrag_bf16(a, PA, 512, arow, kb, klo);
        __builtin_prefetch(PA + (size_t)arow * 512 + kb + 32, 0, 1);  // global_prefetch_b8
        #pragma unroll
        for (int q = 0; q < 4; ++q) {
            ABFrag b;
            load_bfrag_bf16(b, PB, 512, (tg * 4 + q) * 16 + m, kb, hi);
            c[q] = wmma_bf16(a, b, c[q]);
        }
    }
    #pragma unroll
    for (int q = 0; q < 4; ++q) {
        int j  = (tg * 4 + q) * 16 + (lane & 15);
        int kj = tgt[j];
        float bnj = bn[j];
        #pragma unroll
        for (int r = 0; r < 8; ++r) {
            int i = ti * 16 + r + hi * 8;
            float d  = fmaxf(an[i] + bnj - 2.0f * c[q][r], 0.0f);
            float kv = __expf(-0.5f * d);    // gamma = 1/(2*sigma^2), sigma=1
            int ki = tgt[i];
            if (mode == 2)      atomicAdd(&out[(size_t)ki * 1000 + kj], kv);
            else if (ki == kj)  atomicAdd(&out[ki], kv);
        }
    }
}

// ---------------------------------------------------------------------------
// C[k,l] = max(|wt_k|^2+|ws_l|^2-2 wt_k.ws_l, 0) - 0.1*MMD2[k,l]; emits
// Kmat = exp(-C/0.1) and Kmat^T (ld=1024, padded rows finite). 16x64 strips.
// ---------------------------------------------------------------------------
__global__ void elot_build_C(const __bf16* __restrict__ WT, const __bf16* __restrict__ WS,
                             const float* __restrict__ wtn, const float* __restrict__ wsn,
                             const float* __restrict__ stt, const float* __restrict__ sss,
                             const float* __restrict__ Sts, const float* __restrict__ cnt,
                             float* __restrict__ C, float* __restrict__ KM,
                             float* __restrict__ KMT) {
    int wid  = (blockIdx.x * blockDim.x + threadIdx.x) >> 5;
    int lane = threadIdx.x & 31;
    const int TI = 64, NG = 16;              // 1024/16 row tiles, 1024/64 col groups
    if (wid >= TI * NG) return;
    int ti = wid / NG, tg = wid % NG;
    int m = lane & 15, hi = lane >> 4, klo = hi * 8;
    int arow = ti * 16 + m;
    f32x8 c[4];
    #pragma unroll
    for (int q = 0; q < 4; ++q) c[q] = f32x8{0.f, 0.f, 0.f, 0.f, 0.f, 0.f, 0.f, 0.f};
    #pragma unroll 2
    for (int kb = 0; kb < 512; kb += 32) {
        ABFrag a;
        load_afrag_bf16(a, WT, 512, arow, kb, klo);
        #pragma unroll
        for (int q = 0; q < 4; ++q) {
            ABFrag b;
            load_bfrag_bf16(b, WS, 512, (tg * 4 + q) * 16 + m, kb, hi);
            c[q] = wmma_bf16(a, b, c[q]);
        }
    }
    #pragma unroll
    for (int q = 0; q < 4; ++q) {
        int l = (tg * 4 + q) * 16 + (lane & 15);
        float wsl = wsn[l];
        #pragma unroll
        for (int r = 0; r < 8; ++r) {
            int k = ti * 16 + r + hi * 8;
            float cw = fmaxf(wtn[k] + wsl - 2.0f * c[q][r], 0.0f);
            float mmd = 0.0f;
            if (k < 1000 && l < 1000) {
                float ck = cnt[k], cl = cnt[l];
                if (ck > 0.0f && cl > 0.0f)
                    mmd = stt[k] / (ck * ck) + sss[l] / (cl * cl)
                        - 2.0f * Sts[(size_t)k * 1000 + l] / (ck * cl);
            }
            float Cv = cw - 0.1f * mmd;
            C[(size_t)k * 1024 + l]   = Cv;
            float km = __expf(-10.0f * Cv);  // reg = 0.1
            KM[(size_t)k * 1024 + l]  = km;
            KMT[(size_t)l * 1024 + k] = km;
        }
    }
}

__global__ void elot_init_uv(float* __restrict__ U, float* __restrict__ V) {
    int i = blockIdx.x * blockDim.x + threadIdx.x;
    if (i < 1000) { U[i] = 1.0f / 1000.0f; V[i] = 1.0f / 1000.0f; }
}

// One Sinkhorn half-step: y[j] = (1/1000) / dot(Krow[j,:1000], x). Wave/row.
__global__ void elot_sink_step(const float* __restrict__ Krow, const float* __restrict__ x,
                               float* __restrict__ y) {
    int w = (blockIdx.x * blockDim.x + threadIdx.x) >> 5;
    int lane = threadIdx.x & 31;
    if (w >= 1000) return;
    const float* row = Krow + (size_t)w * 1024;
    float s = 0.f;
    for (int i = lane; i < 1000; i += 32) s += row[i] * x[i];
    #pragma unroll
    for (int off = 16; off; off >>= 1) s += __shfl_down(s, off, 32);
    if (lane == 0) y[w] = (1.0f / 1000.0f) / s;
}

__global__ void elot_final(const float* __restrict__ C, const float* __restrict__ KM,
                           const float* __restrict__ u, const float* __restrict__ v,
                           float* __restrict__ out) {
    __shared__ float sm[8];
    float s = 0.f;
    for (int idx = blockIdx.x * blockDim.x + threadIdx.x; idx < 1000 * 1000;
         idx += gridDim.x * blockDim.x) {
        int i = idx / 1000, j = idx - i * 1000;
        float pi = u[i] * KM[(size_t)i * 1024 + j] * v[j];
        s += pi * C[(size_t)i * 1024 + j];
    }
    #pragma unroll
    for (int off = 16; off; off >>= 1) s += __shfl_down(s, off, 32);
    int lane = threadIdx.x & 31, w = threadIdx.x >> 5;
    if (lane == 0) sm[w] = s;
    __syncthreads();
    if (threadIdx.x < 8) {
        s = sm[threadIdx.x];
        #pragma unroll
        for (int off = 4; off; off >>= 1) s += __shfl_down(s, off, 32);
        if (threadIdx.x == 0) atomicAdd(out, s);
    }
}

// ---------------------------------------------------------------------------
extern "C" void kernel_launch(void* const* d_in, const int* in_sizes, int n_in,
                              void* d_out, int out_size, void* d_ws, size_t ws_size,
                              hipStream_t stream) {
    const float* feat_s  = (const float*)d_in[0];   // 4096 x 1024
    const float* feat_t  = (const float*)d_in[1];   // 4096 x 2048
    const float* w_s     = (const float*)d_in[2];   // 1000 x 1024
    const float* w_t     = (const float*)d_in[3];   // 1000 x 2048
    const float* Wt_proj = (const float*)d_in[4];   // 512 x 2048
    const float* Ws_proj = (const float*)d_in[5];   // 512 x 1024
    const int*   target  = (const int*)d_in[6];     // 4096
    (void)in_sizes; (void)n_in; (void)out_size; (void)ws_size;

    char* ws = (char*)d_ws;
    size_t off = 0;
    auto take = [&](size_t bytes) { size_t o = off; off += (bytes + 255) & ~(size_t)255; return o; };

    __bf16* PT  = (__bf16*)(ws + take((size_t)4096 * 512 * 2)); // ft_p bf16
    __bf16* PS  = (__bf16*)(ws + take((size_t)4096 * 512 * 2)); // fs_p bf16
    __bf16* WTP = (__bf16*)(ws + take((size_t)1024 * 512 * 2)); // wt_p bf16 (padded)
    __bf16* WSP = (__bf16*)(ws + take((size_t)1024 * 512 * 2)); // ws_p bf16 (padded)
    float* FTN = (float*)(ws + take(4096 * 4));
    float* FSN = (float*)(ws + take(4096 * 4));
    float* WTN = (float*)(ws + take(1024 * 4));
    float* WSN = (float*)(ws + take(1024 * 4));
    size_t acc0 = off;                                          // zeroed zone
    float* STT = (float*)(ws + take(1024 * 4));
    float* SSS = (float*)(ws + take(1024 * 4));
    float* CNT = (float*)(ws + take(1024 * 4));
    float* STS = (float*)(ws + take((size_t)1000 * 1000 * 4));
    size_t accB = off - acc0;
    float* C   = (float*)(ws + take((size_t)1024 * 1024 * 4));
    float* KM  = (float*)(ws + take((size_t)1024 * 1024 * 4));
    float* KMT = (float*)(ws + take((size_t)1024 * 1024 * 4));
    float* U   = (float*)(ws + take(1024 * 4));
    float* V   = (float*)(ws + take(1024 * 4));

    hipMemsetAsync(ws + acc0, 0, accB, stream);                 // class accumulators
    hipMemsetAsync(d_out, 0, sizeof(float), stream);

    // Projections (WMMA bf16): waves = (Mpad/16)*(512/64); 8 waves/block.
    elot_proj_wmma<<<(256 * 8) / 8, 256, 0, stream>>>(feat_t, Wt_proj, PT, 4096, 4096, 2048);
    elot_proj_wmma<<<(256 * 8) / 8, 256, 0, stream>>>(feat_s, Ws_proj, PS, 4096, 4096, 1024);
    elot_proj_wmma<<<(64 * 8) / 8, 256, 0, stream>>>(w_t, Wt_proj, WTP, 1000, 1024, 2048);
    elot_proj_wmma<<<(64 * 8) / 8, 256, 0, stream>>>(w_s, Ws_proj, WSP, 1000, 1024, 1024);

    elot_rownorm<<<4096 / 8, 256, 0, stream>>>(PT, FTN, 4096);
    elot_rownorm<<<4096 / 8, 256, 0, stream>>>(PS, FSN, 4096);
    elot_rownorm<<<1024 / 8, 256, 0, stream>>>(WTP, WTN, 1024);
    elot_rownorm<<<1024 / 8, 256, 0, stream>>>(WSP, WSN, 1024);

    elot_count<<<4096 / 256, 256, 0, stream>>>(target, CNT, 4096);

    // Fused RBF-kernel + class scatter: waves = 256*64; 8 waves/block.
    elot_dist_cls<<<(256 * 64) / 8, 256, 0, stream>>>(PT, PT, FTN, FTN, target, STT, 0);
    elot_dist_cls<<<(256 * 64) / 8, 256, 0, stream>>>(PS, PS, FSN, FSN, target, SSS, 1);
    elot_dist_cls<<<(256 * 64) / 8, 256, 0, stream>>>(PT, PS, FTN, FSN, target, STS, 2);

    // Cost matrix + Gibbs kernel: waves = 64*16.
    elot_build_C<<<(64 * 16) / 8, 256, 0, stream>>>(WTP, WSP, WTN, WSN, STT, SSS, STS, CNT,
                                                    C, KM, KMT);

    // Sinkhorn: u0 = 1/K; 1000 x { v = b/(K^T u); u = a/(K v) }.
    elot_init_uv<<<4, 256, 0, stream>>>(U, V);
    for (int it = 0; it < 1000; ++it) {
        elot_sink_step<<<125, 256, 0, stream>>>(KMT, U, V);
        elot_sink_step<<<125, 256, 0, stream>>>(KM, V, U);
    }

    elot_final<<<512, 256, 0, stream>>>(C, KM, U, V, (float*)d_out);
}
